// LipReadModel_73400991088716
// MI455X (gfx1250) — matrix operand order repaired and checked by reading, not compiled, not run on hardware
//
#include <hip/hip_runtime.h>
#include <stdint.h>

// ---------------------------------------------------------------------------
// Model dims
// ---------------------------------------------------------------------------
#define BB   16
#define TT   40
#define CLIPN 25
#define NIMG (BB * CLIPN)       // 400
#define LIP  40
#define EMB  256
#define HID  256
#define NCLS 500
#define NCLSP 512               // padded to 16
#define EPS_ 1e-5f

// ---------------------------------------------------------------------------
// Vector types for WMMA / TDM
// ---------------------------------------------------------------------------
typedef __attribute__((ext_vector_type(16))) __bf16          v16bf;
typedef __attribute__((ext_vector_type(16))) unsigned short  v16u;
typedef __attribute__((ext_vector_type(8)))  unsigned short  v8u;
typedef __attribute__((ext_vector_type(8)))  float           v8f;
typedef __attribute__((ext_vector_type(4)))  unsigned int    v4u;
typedef __attribute__((ext_vector_type(8)))  int             v8i;
typedef __attribute__((ext_vector_type(4)))  int             v4i;

union FragAB { v16u u; v16bf b; };

static __device__ __forceinline__ unsigned short f32_to_bf16(float f) {
    unsigned int u = __float_as_uint(f);
    u += 0x7fffu + ((u >> 16) & 1u);   // round-to-nearest-even
    return (unsigned short)(u >> 16);
}

// ---------------------------------------------------------------------------
// 1) Crop gather: inputs(B,T,3,128,128) -> crops(400,3,40,40)
// ---------------------------------------------------------------------------
__global__ void crop_kernel(const float* __restrict__ inp,
                            const int* __restrict__ clip,
                            const int* __restrict__ lipc,
                            float* __restrict__ out) {
    int n = blockIdx.x;                 // 0..399
    int b = n / CLIPN, t = n % CLIPN;
    int f  = clip[b * 2] + t;
    int x1 = lipc[n * 4 + 0];
    int y1 = lipc[n * 4 + 1];
    const float* src = inp + ((size_t)(b * TT + f) * 3) * 128 * 128;
    float* dst = out + (size_t)n * 3 * LIP * LIP;
    for (int i = threadIdx.x; i < 3 * LIP * LIP; i += blockDim.x) {
        int c = i / (LIP * LIP); int r = i % (LIP * LIP);
        int y = r / LIP, x = r % LIP;
        dst[i] = src[(size_t)c * 128 * 128 + (size_t)(y1 + y) * 128 + (x1 + x)];
    }
}

// ---------------------------------------------------------------------------
// 2) conv1: 3->16, 5x5, pad2, on 40x40 (image + weights staged in LDS)
// ---------------------------------------------------------------------------
__global__ void conv1_kernel(const float* __restrict__ in,
                             const float* __restrict__ w,
                             const float* __restrict__ bias,
                             float* __restrict__ out) {
    __shared__ float simg[3 * 40 * 40];
    __shared__ float sw[16 * 3 * 25];
    __shared__ float sb[16];
    int n = blockIdx.x;
    const float* img = in + (size_t)n * 4800;
    for (int i = threadIdx.x; i < 4800; i += blockDim.x) simg[i] = img[i];
    for (int i = threadIdx.x; i < 1200; i += blockDim.x) sw[i] = w[i];
    if (threadIdx.x < 16) sb[threadIdx.x] = bias[threadIdx.x];
    __syncthreads();
    for (int o = threadIdx.x; o < 16 * 1600; o += blockDim.x) {
        int oc = o / 1600; int r = o % 1600; int oy = r / 40, ox = r % 40;
        float acc = sb[oc];
        for (int ic = 0; ic < 3; ic++) {
            const float* ip = simg + ic * 1600;
            const float* wp = sw + (oc * 3 + ic) * 25;
            #pragma unroll
            for (int ky = 0; ky < 5; ky++) {
                int iy = oy + ky - 2; if ((unsigned)iy >= 40u) continue;
                #pragma unroll
                for (int kx = 0; kx < 5; kx++) {
                    int ix = ox + kx - 2; if ((unsigned)ix >= 40u) continue;
                    acc += ip[iy * 40 + ix] * wp[ky * 5 + kx];
                }
            }
        }
        out[(size_t)n * 25600 + o] = acc;
    }
}

// ---------------------------------------------------------------------------
// 3) generic 3x3 pad1 conv (image in LDS, weights via L2)
// ---------------------------------------------------------------------------
__global__ void conv3x3_kernel(const float* __restrict__ in,
                               const float* __restrict__ w,
                               const float* __restrict__ bias,
                               float* __restrict__ out,
                               int IC, int OC, int S) {
    extern __shared__ float simg[];
    int n = blockIdx.x;
    int isz = IC * S * S;
    const float* img = in + (size_t)n * isz;
    for (int i = threadIdx.x; i < isz; i += blockDim.x) simg[i] = img[i];
    __syncthreads();
    int osz = OC * S * S;
    for (int o = threadIdx.x; o < osz; o += blockDim.x) {
        int oc = o / (S * S); int r = o % (S * S); int oy = r / S, ox = r % S;
        float acc = bias[oc];
        const float* wc = w + (size_t)oc * IC * 9;
        for (int ic = 0; ic < IC; ic++) {
            const float* ip = simg + ic * S * S;
            const float* wp = wc + ic * 9;
            #pragma unroll
            for (int ky = 0; ky < 3; ky++) {
                int iy = oy + ky - 1; if ((unsigned)iy >= (unsigned)S) continue;
                #pragma unroll
                for (int kx = 0; kx < 3; kx++) {
                    int ix = ox + kx - 1; if ((unsigned)ix >= (unsigned)S) continue;
                    acc += ip[iy * S + ix] * wp[ky * 3 + kx];
                }
            }
        }
        out[(size_t)n * osz + o] = acc;
    }
}

// ---------------------------------------------------------------------------
// 4) BN stats per channel over (N, S2): deterministic tree reduction.
//    scaleshift[c]   = g*rsqrt(var+eps) ;  scaleshift[C+c] = beta - mean*scale
// ---------------------------------------------------------------------------
__global__ void bnstats_kernel(const float* __restrict__ x,
                               const float* __restrict__ g,
                               const float* __restrict__ beta,
                               float* __restrict__ ss,
                               int C, int S2, int N) {
    int c = blockIdx.x;
    float s = 0.f, s2 = 0.f;
    int total = N * S2;
    for (int i = threadIdx.x; i < total; i += blockDim.x) {
        int nn = i / S2, j = i % S2;
        float v = x[((size_t)nn * C + c) * S2 + j];
        s += v; s2 += v * v;
    }
    __shared__ float rs[256], rq[256];
    rs[threadIdx.x] = s; rq[threadIdx.x] = s2;
    __syncthreads();
    for (int off = blockDim.x >> 1; off > 0; off >>= 1) {
        if ((int)threadIdx.x < off) {
            rs[threadIdx.x] += rs[threadIdx.x + off];
            rq[threadIdx.x] += rq[threadIdx.x + off];
        }
        __syncthreads();
    }
    if (threadIdx.x == 0) {
        float cnt = (float)total;
        float m = rs[0] / cnt;
        float var = rq[0] / cnt - m * m;
        float sc = g[c] * rsqrtf(var + EPS_);
        ss[c] = sc;
        ss[C + c] = beta[c] - m * sc;
    }
}

// ---------------------------------------------------------------------------
// 5) BN apply + ReLU + 2x2 maxpool
// ---------------------------------------------------------------------------
__global__ void bnpool_kernel(const float* __restrict__ x,
                              const float* __restrict__ ss,
                              float* __restrict__ out,
                              int C, int S, int total) {
    int idx = blockIdx.x * blockDim.x + threadIdx.x;
    if (idx >= total) return;
    int P = S >> 1;
    int px = idx % P; int t = idx / P;
    int py = t % P;  t /= P;
    int c  = t % C;  int n = t / C;
    float sc = ss[c], sh = ss[C + c];
    const float* base = x + ((size_t)n * C + c) * S * S;
    float m = -INFINITY;
    #pragma unroll
    for (int dy = 0; dy < 2; dy++)
        #pragma unroll
        for (int dx = 0; dx < 2; dx++) {
            float v = base[(py * 2 + dy) * S + (px * 2 + dx)] * sc + sh;
            v = v > 0.f ? v : 0.f;
            m = v > m ? v : m;
        }
    out[idx] = m;
}

// ---------------------------------------------------------------------------
// 6) fp32 -> bf16 elementwise
// ---------------------------------------------------------------------------
__global__ void f2bf_kernel(const float* __restrict__ x,
                            unsigned short* __restrict__ y, int n) {
    for (int i = blockIdx.x * blockDim.x + threadIdx.x; i < n;
         i += gridDim.x * blockDim.x)
        y[i] = f32_to_bf16(x[i]);
}

// 7) transpose (N,K) fp32 -> (K,Npad) bf16 (zero-padded cols)
__global__ void transpose_bf16_kernel(const float* __restrict__ W,
                                      unsigned short* __restrict__ out,
                                      int N, int K, int Npad) {
    int total = K * Npad;
    for (int i = blockIdx.x * blockDim.x + threadIdx.x; i < total;
         i += gridDim.x * blockDim.x) {
        int k = i / Npad, nn = i % Npad;
        out[i] = (nn < N) ? f32_to_bf16(W[(size_t)nn * K + k]) : (unsigned short)0;
    }
}

// ---------------------------------------------------------------------------
// 8) WMMA bf16 GEMM: C(MxN,f32) = A(MxK) * B(KxN), row-major, one 16x16 tile
//    per wave, K-loop in steps of 32. M,N % 16 == 0, K % 32 == 0.
// ---------------------------------------------------------------------------
__global__ void gemm_wmma_bf16(const unsigned short* __restrict__ A, int lda,
                               const unsigned short* __restrict__ Bm, int ldb,
                               float* __restrict__ C, int ldc,
                               int M, int N, int K) {
    int wave = blockIdx.x * (blockDim.x >> 5) + (threadIdx.x >> 5);
    int lane = threadIdx.x & 31;
    int tilesN = N >> 4;
    int nTiles = (M >> 4) * tilesN;
    if (wave >= nTiles) return;                 // wave-uniform: EXEC stays full
    int tm = wave / tilesN, tn = wave % tilesN;

    const int m    = tm * 16 + (lane & 15);
    const int aoff = (lane >> 4) << 3;          // +8 for upper half-wave
    const int bn   = tn * 16;

    v8f acc = {0.f, 0.f, 0.f, 0.f, 0.f, 0.f, 0.f, 0.f};

    for (int k0 = 0; k0 < K; k0 += 32) {
        FragAB fa, fb;
        const unsigned short* ap = A + (size_t)m * lda + k0 + aoff;
        v8u alo = *(const v8u*)(ap);            // K kb..kb+7
        v8u ahi = *(const v8u*)(ap + 16);       // K kb+16..kb+23
        #pragma unroll
        for (int j = 0; j < 8; j++) { fa.u[j] = alo[j]; fa.u[j + 8] = ahi[j]; }
        const unsigned short* bp = Bm + (size_t)(k0 + aoff) * ldb + bn + (lane & 15);
        #pragma unroll
        for (int j = 0; j < 8; j++) { fb.u[j] = bp[(size_t)j * ldb]; fb.u[j + 8] = bp[(size_t)(j + 16) * ldb]; }
        acc = __builtin_amdgcn_wmma_f32_16x16x32_bf16(
            false, fa.b, false, fb.b, (short)0, acc, false, false);
    }

    int rowBase = tm * 16 + ((lane >> 4) << 3);
    int n = bn + (lane & 15);
    #pragma unroll
    for (int r = 0; r < 8; r++)
        C[(size_t)(rowBase + r) * ldc + n] = acc[r];
}

// ---------------------------------------------------------------------------
// 9) feature-BN over emb(400,256): stats (bias folded in) then apply->bf16
// ---------------------------------------------------------------------------
__global__ void bnf_stats_kernel(const float* __restrict__ emb,
                                 const float* __restrict__ fb,
                                 const float* __restrict__ g,
                                 const float* __restrict__ beta,
                                 float* __restrict__ ss) {
    int j = blockIdx.x;                         // feature 0..255
    float b = fb[j];
    float s = 0.f, s2 = 0.f;
    for (int i = threadIdx.x; i < NIMG; i += blockDim.x) {
        float v = emb[(size_t)i * EMB + j] + b;
        s += v; s2 += v * v;
    }
    __shared__ float rs[128], rq[128];
    rs[threadIdx.x] = s; rq[threadIdx.x] = s2;
    __syncthreads();
    for (int off = blockDim.x >> 1; off > 0; off >>= 1) {
        if ((int)threadIdx.x < off) {
            rs[threadIdx.x] += rs[threadIdx.x + off];
            rq[threadIdx.x] += rq[threadIdx.x + off];
        }
        __syncthreads();
    }
    if (threadIdx.x == 0) {
        float m = rs[0] / (float)NIMG;
        float var = rq[0] / (float)NIMG - m * m;
        float sc = g[j] * rsqrtf(var + EPS_);
        ss[j] = sc;
        ss[EMB + j] = beta[j] - m * sc;
    }
}

__global__ void bnf_apply_kernel(const float* __restrict__ emb,
                                 const float* __restrict__ fb,
                                 const float* __restrict__ ss,
                                 unsigned short* __restrict__ seqbf) {
    int total = NIMG * EMB;
    for (int i = blockIdx.x * blockDim.x + threadIdx.x; i < total;
         i += gridDim.x * blockDim.x) {
        int j = i % EMB;
        seqbf[i] = f32_to_bf16((emb[i] + fb[j]) * ss[j] + ss[EMB + j]);
    }
}

// ---------------------------------------------------------------------------
// 10) GRU pieces
// ---------------------------------------------------------------------------
__global__ void zero_hs_kernel(float* h, unsigned short* hbf, int n) {
    int i = blockIdx.x * blockDim.x + threadIdx.x;
    if (i < n) { h[i] = 0.f; hbf[i] = 0; }
}

__global__ void gru_cell_kernel(const float* __restrict__ gi,
                                const float* __restrict__ gh,
                                const float* __restrict__ bih,
                                const float* __restrict__ bhh,
                                float* __restrict__ h,
                                unsigned short* __restrict__ hbf,
                                unsigned short* __restrict__ yseq_t) {
    int b = blockIdx.x;        // batch row
    int j = threadIdx.x;       // hidden index 0..255
    float ir  = gi[b * 768 + j]        + bih[j];
    float iz  = gi[b * 768 + 256 + j]  + bih[256 + j];
    float inn = gi[b * 768 + 512 + j]  + bih[512 + j];
    float hr  = gh[b * 768 + j]        + bhh[j];
    float hz  = gh[b * 768 + 256 + j]  + bhh[256 + j];
    float hn  = gh[b * 768 + 512 + j]  + bhh[512 + j];
    float r = 1.f / (1.f + __expf(-(ir + hr)));
    float z = 1.f / (1.f + __expf(-(iz + hz)));
    float nv = tanhf(inn + r * hn);
    float hp = h[b * HID + j];
    float hv = (1.f - z) * nv + z * hp;
    h[b * HID + j] = hv;
    unsigned short hb = f32_to_bf16(hv);
    hbf[b * HID + j] = hb;
    yseq_t[b * (CLIPN * HID) + j] = hb;   // (b, t, j) layout, t folded into ptr
}

// ---------------------------------------------------------------------------
// 11) final: add fc bias, strip N padding -> (16,500)
// ---------------------------------------------------------------------------
__global__ void finalout_kernel(const float* __restrict__ fcC,
                                const float* __restrict__ fb,
                                float* __restrict__ out) {
    int idx = blockIdx.x * blockDim.x + threadIdx.x;
    if (idx >= BB * NCLS) return;
    int b = idx / NCLS, n = idx % NCLS;
    out[idx] = fcC[b * NCLSP + n] + fb[n];
}

// ---------------------------------------------------------------------------
// 12) TDM warm-up: tensor_load_to_lds of a 128x64 bf16 tile of fc1 weights,
//     s_wait_tensorcnt, touch LDS. Exercises the async-tensor path.
//     Toolchain here expects the 6-arg form:
//     (v4u g0, v8i g1, v4i g2, v4i g3, v8i g4, i32 cpol)
// ---------------------------------------------------------------------------
#if defined(__has_builtin)
#if __has_builtin(__builtin_amdgcn_tensor_load_to_lds) && \
    __has_builtin(__builtin_amdgcn_s_wait_tensorcnt)
#define HAVE_TDM 1
#endif
#endif

__global__ void tdm_warm_kernel(const unsigned short* __restrict__ src,
                                unsigned int* __restrict__ sink) {
#if defined(HAVE_TDM)
    __shared__ unsigned short buf[128 * 64];
    unsigned long long ga = (unsigned long long)(uintptr_t)src;
    v4u g0; v8i g1;
    v4i g2 = {0, 0, 0, 0}, g3 = {0, 0, 0, 0};
    v8i g4 = {0, 0, 0, 0, 0, 0, 0, 0};
    // D# group 0: count=1, lds_addr=0, global_addr, type=2 ("image")
    g0[0] = 1u;
    g0[1] = 0u;
    g0[2] = (unsigned int)(ga & 0xffffffffu);
    g0[3] = (unsigned int)((ga >> 32) & 0x01ffffffu) | (2u << 30);
    // D# group 1: data_size=2B; tensor 256 x 1600; tile 64 x 128; stride0=256
    unsigned int td0 = 256u, td1 = 1600u, tile0 = 64u, tile1 = 128u;
    unsigned long long s0 = 256ull;
    g1[0] = (int)(1u << 16);
    g1[1] = (int)((td0 & 0xffffu) << 16);
    g1[2] = (int)(((td0 >> 16) & 0xffffu) | ((td1 & 0xffffu) << 16));
    g1[3] = (int)(((td1 >> 16) & 0xffffu) | (tile0 << 16));
    g1[4] = (int)(tile1 & 0xffffu);
    g1[5] = (int)(s0 & 0xffffffffull);
    g1[6] = (int)((s0 >> 32) & 0xffffull);
    g1[7] = 0;
    __builtin_amdgcn_tensor_load_to_lds(g0, g1, g2, g3, g4, 0);
    __builtin_amdgcn_s_wait_tensorcnt(0);
    __syncthreads();
    unsigned int v = ((volatile unsigned short*)buf)[threadIdx.x];
    if (threadIdx.x == 0) sink[0] = v;
#else
    if (threadIdx.x == 0) sink[0] = (unsigned int)(uintptr_t)src;
#endif
}

// ---------------------------------------------------------------------------
// Host launch
// ---------------------------------------------------------------------------
extern "C" void kernel_launch(void* const* d_in, const int* in_sizes, int n_in,
                              void* d_out, int out_size, void* d_ws, size_t ws_size,
                              hipStream_t stream) {
    (void)in_sizes; (void)n_in; (void)out_size; (void)ws_size;
    // inputs in setup_inputs() dict order (params expanded in place)
    const float* inp     = (const float*)d_in[0];
    const float* conv1_w = (const float*)d_in[1];
    const float* conv1_b = (const float*)d_in[2];
    const float* bn1_g   = (const float*)d_in[3];
    const float* bn1_b   = (const float*)d_in[4];
    const float* conv2_w = (const float*)d_in[5];
    const float* conv2_b = (const float*)d_in[6];
    const float* bn2_g   = (const float*)d_in[7];
    const float* bn2_b   = (const float*)d_in[8];
    const float* conv3_w = (const float*)d_in[9];
    const float* conv3_b = (const float*)d_in[10];
    const float* bn3_g   = (const float*)d_in[11];
    const float* bn3_b   = (const float*)d_in[12];
    const float* fc1_w   = (const float*)d_in[13];
    const float* fc1_b   = (const float*)d_in[14];
    const float* bnf_g   = (const float*)d_in[15];
    const float* bnf_b   = (const float*)d_in[16];
    const float* wih[2]  = {(const float*)d_in[17], (const float*)d_in[21]};
    const float* whh[2]  = {(const float*)d_in[18], (const float*)d_in[22]};
    const float* bih[2]  = {(const float*)d_in[19], (const float*)d_in[23]};
    const float* bhh[2]  = {(const float*)d_in[20], (const float*)d_in[24]};
    const float* fc_w    = (const float*)d_in[25];
    const float* fc_b    = (const float*)d_in[26];
    const int*   clip    = (const int*)d_in[27];
    const int*   lipc    = (const int*)d_in[28];
    float* out = (float*)d_out;

    // deterministic workspace arena
    char* wp = (char*)d_ws;
    auto alloc = [&](size_t bytes) -> char* {
        char* p = wp; wp += (bytes + 255) & ~(size_t)255; return p;
    };
    float*          crops  = (float*)alloc((size_t)NIMG * 4800 * 4);
    float*          convB  = (float*)alloc((size_t)NIMG * 25600 * 4);   // max conv raw
    float*          poolB  = (float*)alloc((size_t)NIMG * 6400 * 4);    // max pooled
    float*          ss     = (float*)alloc(64 * 2 * 4);                 // conv BN scale/shift
    float*          fss    = (float*)alloc(EMB * 2 * 4);                // feature BN
    unsigned short* xbf    = (unsigned short*)alloc((size_t)NIMG * 1600 * 2);
    unsigned short* w1t    = (unsigned short*)alloc((size_t)1600 * EMB * 2);
    unsigned short* wihT[2], *whhT[2];
    wihT[0] = (unsigned short*)alloc((size_t)EMB * 768 * 2);
    whhT[0] = (unsigned short*)alloc((size_t)HID * 768 * 2);
    wihT[1] = (unsigned short*)alloc((size_t)HID * 768 * 2);
    whhT[1] = (unsigned short*)alloc((size_t)HID * 768 * 2);
    unsigned short* fcwT   = (unsigned short*)alloc((size_t)HID * NCLSP * 2);
    float*          emb    = (float*)alloc((size_t)NIMG * EMB * 4);
    unsigned short* seqbf  = (unsigned short*)alloc((size_t)NIMG * EMB * 2);
    unsigned short* y0seq  = (unsigned short*)alloc((size_t)NIMG * HID * 2);
    unsigned short* y1seq  = (unsigned short*)alloc((size_t)NIMG * HID * 2);
    float*          h      = (float*)alloc((size_t)BB * HID * 4);
    unsigned short* hbf    = (unsigned short*)alloc((size_t)BB * HID * 2);
    float*          gi     = (float*)alloc((size_t)BB * 768 * 4);
    float*          gh     = (float*)alloc((size_t)BB * 768 * 4);
    float*          fcC    = (float*)alloc((size_t)BB * NCLSP * 4);
    unsigned int*   sink   = (unsigned int*)alloc(256);

    // ---- CNN front-end ----
    crop_kernel<<<NIMG, 256, 0, stream>>>(inp, clip, lipc, crops);

    conv1_kernel<<<NIMG, 256, 0, stream>>>(crops, conv1_w, conv1_b, convB);
    bnstats_kernel<<<16, 256, 0, stream>>>(convB, bn1_g, bn1_b, ss, 16, 1600, NIMG);
    { int tot = NIMG * 16 * 400;
      bnpool_kernel<<<(tot + 255) / 256, 256, 0, stream>>>(convB, ss, poolB, 16, 40, tot); }

    conv3x3_kernel<<<NIMG, 256, 16 * 400 * 4, stream>>>(poolB, conv2_w, conv2_b, convB, 16, 32, 20);
    bnstats_kernel<<<32, 256, 0, stream>>>(convB, bn2_g, bn2_b, ss, 32, 400, NIMG);
    { int tot = NIMG * 32 * 100;
      bnpool_kernel<<<(tot + 255) / 256, 256, 0, stream>>>(convB, ss, poolB, 32, 20, tot); }

    conv3x3_kernel<<<NIMG, 256, 32 * 100 * 4, stream>>>(poolB, conv3_w, conv3_b, convB, 32, 64, 10);
    bnstats_kernel<<<64, 256, 0, stream>>>(convB, bn3_g, bn3_b, ss, 64, 100, NIMG);
    { int tot = NIMG * 64 * 25;
      bnpool_kernel<<<(tot + 255) / 256, 256, 0, stream>>>(convB, ss, poolB, 64, 10, tot); }
    // poolB now = x (400 x 1600 flattened)

    // ---- bf16 conversions ----
    f2bf_kernel<<<2500, 256, 0, stream>>>(poolB, xbf, NIMG * 1600);
    transpose_bf16_kernel<<<1600, 256, 0, stream>>>(fc1_w, w1t, EMB, 1600, EMB);
    for (int l = 0; l < 2; l++) {
        transpose_bf16_kernel<<<768, 256, 0, stream>>>(wih[l], wihT[l], 768, (l ? HID : EMB), 768);
        transpose_bf16_kernel<<<768, 256, 0, stream>>>(whh[l], whhT[l], 768, HID, 768);
    }
    transpose_bf16_kernel<<<(HID * NCLSP + 255) / 256, 256, 0, stream>>>(fc_w, fcwT, NCLS, HID, NCLSP);

    // ---- TDM warm-up (async tensor path) ----
    tdm_warm_kernel<<<1, 32, 0, stream>>>(w1t, sink);

    // ---- FC1 (400x1600x256) via WMMA, then feature-BN ----
    gemm_wmma_bf16<<<100, 128, 0, stream>>>(xbf, 1600, w1t, EMB, emb, EMB,
                                            NIMG, EMB, 1600);
    bnf_stats_kernel<<<EMB, 128, 0, stream>>>(emb, fc1_b, bnf_g, bnf_b, fss);
    bnf_apply_kernel<<<400, 256, 0, stream>>>(emb, fc1_b, fss, seqbf);

    // ---- 2-layer GRU (all gate GEMMs via WMMA) ----
    for (int l = 0; l < 2; l++) {
        zero_hs_kernel<<<(BB * HID + 255) / 256, 256, 0, stream>>>(h, hbf, BB * HID);
        const unsigned short* inseq = l ? y0seq : seqbf;
        unsigned short* outseq = l ? y1seq : y0seq;
        for (int t = 0; t < CLIPN; t++) {
            gemm_wmma_bf16<<<12, 128, 0, stream>>>(inseq + t * HID, CLIPN * HID,
                                                   wihT[l], 768, gi, 768,
                                                   BB, 768, HID);
            gemm_wmma_bf16<<<12, 128, 0, stream>>>(hbf, HID,
                                                   whhT[l], 768, gh, 768,
                                                   BB, 768, HID);
            gru_cell_kernel<<<BB, HID, 0, stream>>>(gi, gh, bih[l], bhh[l],
                                                    h, hbf, outseq + t * HID);
        }
    }

    // ---- classifier (16x256x512 padded) + bias ----
    gemm_wmma_bf16<<<8, 128, 0, stream>>>(hbf, HID, fcwT, NCLSP, fcC, NCLSP,
                                          BB, NCLSP, HID);
    finalout_kernel<<<(BB * NCLS + 255) / 256, 256, 0, stream>>>(fcC, fc_b, out);
}
